// SW_MSA_17420387352624
// MI455X (gfx1250) — matrix operand order repaired and checked by reading, not compiled, not run on hardware
//
#include <hip/hip_runtime.h>
#include <hip/hip_bf16.h>

typedef __bf16 bf16;
typedef __attribute__((ext_vector_type(16))) __bf16 v16bf;
typedef __attribute__((ext_vector_type(8)))  float  v8f;

union FragBF { v16bf v; __bf16 e[16]; uint4 q[2]; };
union AccF   { v8f   v; float  e[8]; };

__device__ __forceinline__ void zero_acc(AccF& a) {
    for (int i = 0; i < 8; ++i) a.e[i] = 0.0f;
}

__device__ __forceinline__ v8f wmma_bf16(v16bf a, v16bf b, v8f c) {
    return __builtin_amdgcn_wmma_f32_16x16x32_bf16(false, a, false, b, (short)0, c, false, false);
}

// A fragment: 16x32 bf16 tile rows [row0, row0+16), K base k0, zero for K>=kmax.
// lane L: m = L&15, b0 = 8*(L>>4); e<8 -> K=k0+b0+e ; e>=8 -> K=k0+16+b0+(e-8)
__device__ __forceinline__ v16bf fragA(const bf16* t, int ld, int row0, int k0, int kmax) {
    const int lane = threadIdx.x & 31;
    const int r = lane & 15;
    const int b0 = (lane >> 4) << 3;
    const bf16* rp = t + (row0 + r) * ld;
    FragBF f;
    const int kA = k0 + b0;
    const int kB = k0 + 16 + b0;
    if (kA + 7 < kmax) {
        f.q[0] = *(const uint4*)(rp + kA);
    } else {
        for (int e = 0; e < 8; ++e) { int k = kA + e; f.e[e] = (k < kmax) ? rp[k] : (bf16)0.0f; }
    }
    if (kB + 7 < kmax) {
        f.q[1] = *(const uint4*)(rp + kB);
    } else {
        for (int e = 0; e < 8; ++e) { int k = kB + e; f.e[8 + e] = (k < kmax) ? rp[k] : (bf16)0.0f; }
    }
    return f.v;
}

// B fragment: B is K x N (32x16). Source tile stored N-major (row n holds K values).
// lane L: n = L&15, g = L>>4; e<8 -> K = k0 + 8*g + e ; e>=8 -> K = k0 + 16 + 8*g + (e-8)
__device__ __forceinline__ v16bf fragB(const bf16* t, int ld, int row0, int k0, int kmax) {
    const int lane = threadIdx.x & 31;
    const int n = lane & 15;
    const int g = lane >> 4;
    const bf16* rp = t + (row0 + n) * ld;
    FragBF f;
    const int kA = k0 + (g << 3);
    if (kA + 23 < kmax) {
        f.q[0] = *(const uint4*)(rp + kA);
        f.q[1] = *(const uint4*)(rp + kA + 16);
    } else {
        for (int e = 0; e < 16; ++e) { int k = kA + e + ((e >= 8) ? 8 : 0); f.e[e] = (k < kmax) ? rp[k] : (bf16)0.0f; }
    }
    return f.v;
}

// ---------------- kernel 0: weight conversion f32 -> bf16 ----------------
__global__ void cvt_kernel(const float* __restrict__ wq, const float* __restrict__ wp,
                           bf16* __restrict__ wqb, bf16* __restrict__ wpb) {
    int i = blockIdx.x * 256 + threadIdx.x;
    if (i < 384 * 1152) wqb[i] = (bf16)wq[i];
    if (i < 384 * 384)  wpb[i] = (bf16)wp[i];
}

// ---------------- kernel 1: fused roll+partition+QKV GEMM ----------------
// C(73728 x 1152) = gather(x)(73728 x 384) @ Wqkv(384 x 1152) + b
// scatter into q[mw,h][v][d], k[mw,h][v][d], vT[mw,h][d][v] (bf16)
// Double-buffered LDS; per-thread gather addresses hoisted out of the K loop.
__global__ __launch_bounds__(128) void qkv_kernel(
    const float* __restrict__ x, const bf16* __restrict__ wq, const float* __restrict__ bq,
    bf16* __restrict__ qg, bf16* __restrict__ kg, bf16* __restrict__ vtg) {
    __shared__ __align__(16) bf16 As[2][64 * 32];   // row-major, ld=32
    __shared__ __align__(16) bf16 Bs[2][64 * 32];   // N-major: Bs[n][k], ld=32
    const int tid = threadIdx.x;
    const int lane = tid & 31, wave = tid >> 5;
    const int m0 = blockIdx.x * 64, n0 = blockIdx.y * 64;
    const int wm = (wave >> 1) * 32, wn = (wave & 1) * 32;

    // ---- hoisted A-gather addresses (roll + window partition), invariant in K ----
    const float* asrc[4];
    int aoff[4];
#pragma unroll
    for (int t = 0; t < 4; ++t) {
        int c = tid + t * 128;                 // 0..511
        int row = c >> 3, col = (c & 7) << 2;  // 4 floats per chunk
        int m = m0 + row;
        int mw = m / 36, vv = m - mw * 36;
        int b = mw >> 6, wiw = mw & 63;
        int hr = (wiw >> 3) * 6 + vv / 6;
        int wr = (wiw & 7) * 6 + vv % 6;
        int hs = hr + 3; if (hs >= 48) hs -= 48;
        int wsrc = wr + 3; if (wsrc >= 48) wsrc -= 48;
        asrc[t] = x + (((size_t)(b * 48 + hs) * 48 + wsrc) * 384 + col);
        aoff[t] = row * 32 + col;
    }
    // ---- hoisted B-source addresses ----
    const bf16* bsrc[2];
    int bn[2], bk[2];
#pragma unroll
    for (int t = 0; t < 2; ++t) {
        int c = tid + t * 128;                 // 0..255
        int kr = c >> 3, np = (c & 7) << 3;
        bsrc[t] = wq + (size_t)kr * 1152 + n0 + np;
        bn[t] = np; bk[t] = kr;
    }

    // load-all then store-all so global loads stay in flight together
    auto fill = [&](int buf, int k0) {
        float4 fa[4];
        uint4  pb[2];
#pragma unroll
        for (int t = 0; t < 4; ++t) fa[t] = *(const float4*)(asrc[t] + k0);
#pragma unroll
        for (int t = 0; t < 2; ++t) {
            const bf16* src = bsrc[t] + (size_t)k0 * 1152;
            if (k0 + 32 < 384) __builtin_prefetch(src + 32 * 1152, 0, 0);
            pb[t] = *(const uint4*)src;
        }
#pragma unroll
        for (int t = 0; t < 4; ++t) {
            bf16* d = &As[buf][aoff[t]];
            d[0] = (bf16)fa[t].x; d[1] = (bf16)fa[t].y;
            d[2] = (bf16)fa[t].z; d[3] = (bf16)fa[t].w;
        }
#pragma unroll
        for (int t = 0; t < 2; ++t) {
            const bf16* pe = (const bf16*)&pb[t];
            for (int e = 0; e < 8; ++e) Bs[buf][(bn[t] + e) * 32 + bk[t]] = pe[e];
        }
    };

    AccF acc[2][2];
    zero_acc(acc[0][0]); zero_acc(acc[0][1]); zero_acc(acc[1][0]); zero_acc(acc[1][1]);

    fill(0, 0);
    __syncthreads();
    for (int kt = 0; kt < 12; ++kt) {
        int cur = kt & 1;
        if (kt < 11) fill(cur ^ 1, (kt + 1) * 32);
        v16bf bfr[2];
        bfr[0] = fragB(Bs[cur], 32, wn, 0, 32);
        bfr[1] = fragB(Bs[cur], 32, wn + 16, 0, 32);
        for (int fm = 0; fm < 2; ++fm) {
            v16bf a = fragA(As[cur], 32, wm + fm * 16, 0, 32);
            acc[fm][0].v = wmma_bf16(a, bfr[0], acc[fm][0].v);
            acc[fm][1].v = wmma_bf16(a, bfr[1], acc[fm][1].v);
        }
        __syncthreads();
    }

    const int nl = lane & 15, mh = (lane >> 4) << 3;
    for (int fm = 0; fm < 2; ++fm)
        for (int fn = 0; fn < 2; ++fn)
            for (int i = 0; i < 8; ++i) {
                int m = m0 + wm + fm * 16 + mh + i;
                int n = n0 + wn + fn * 16 + nl;
                float val = acc[fm][fn].e[i] + bq[n];
                int which = n / 384, nr = n - which * 384;
                int hh = nr / 48, d = nr - hh * 48;
                int mw = m / 36, vv = m - mw * 36;
                size_t hb = (size_t)(mw * 8 + hh);
                bf16 o = (bf16)val;
                if (which == 0)      qg[hb * (36 * 48) + vv * 48 + d] = o;
                else if (which == 1) kg[hb * (36 * 48) + vv * 48 + d] = o;
                else                 vtg[hb * (48 * 36) + d * 36 + vv] = o;
            }
}

// ---------------- kernel 2: fused masked attention (one wave per head) ----------------
__global__ __launch_bounds__(64) void attn_kernel(
    const bf16* __restrict__ qg, const bf16* __restrict__ kg,
    const bf16* __restrict__ vtg, bf16* __restrict__ yg) {
    __shared__ __align__(16) char smem[2 * 27648];
    const int wave = threadIdx.x >> 5, lane = threadIdx.x & 31;
    const int mw = blockIdx.x;
    const int h = blockIdx.y * 2 + wave;

    char* base = smem + wave * 27648;
    bf16*  qt = (bf16*)base;            // 48x48 (rows 0..35 valid)
    bf16*  kt = qt + 48 * 48;           // 48x48 (rows 0..35 valid)
    bf16*  vt = kt + 48 * 48;           // 48x48 (cols 0..35 valid)
    bf16*  at = vt + 48 * 48;           // attn 48x48 bf16
    float* sc = (float*)(at + 48 * 48); // scores 48x48 f32

    const size_t tq = (size_t)(mw * 8 + h) * (36 * 48);
    const size_t tv = (size_t)(mw * 8 + h) * (48 * 36);
    {
        const uint4* sq = (const uint4*)(qg + tq);
        const uint4* sk = (const uint4*)(kg + tq);
        uint4* dq = (uint4*)qt;
        uint4* dk = (uint4*)kt;
        for (int c = lane; c < 216; c += 32) { dq[c] = sq[c]; dk[c] = sk[c]; }
        const uint2* sv = (const uint2*)(vtg + tv);
        for (int c = lane; c < 432; c += 32) {
            int row = c / 9, p = c - row * 9;
            *(uint2*)(vt + row * 48 + p * 4) = sv[c];
        }
    }
    __syncthreads();

    // scores = q @ k^T * scale   (K = 48 -> ksteps of 32 + 16)
    const float scale = 0.14433756729740643f;  // 48^-0.5
    const int nl = lane & 15, mh = (lane >> 4) << 3;
    for (int mi = 0; mi < 3; ++mi) {
        v16bf a0 = fragA(qt, 48, mi * 16, 0, 48);
        v16bf a1 = fragA(qt, 48, mi * 16, 32, 48);
        for (int ni = 0; ni < 3; ++ni) {
            AccF acc; zero_acc(acc);
            acc.v = wmma_bf16(a0, fragB(kt, 48, ni * 16, 0, 48), acc.v);
            acc.v = wmma_bf16(a1, fragB(kt, 48, ni * 16, 32, 48), acc.v);
            for (int i = 0; i < 8; ++i)
                sc[(mi * 16 + mh + i) * 48 + ni * 16 + nl] = acc.e[i] * scale;
        }
    }
    __syncthreads();

    // shift mask + softmax, row per lane. region id from window position.
    const int wiw = mw & 63;
    const int wh6 = (wiw >> 3) * 6, ww6 = (wiw & 7) * 6;
    for (int v = lane; v < 36; v += 32) {
        int rv = (((wh6 + v / 6) < 45) ? 2 : 0) | (((ww6 + v % 6) < 45) ? 1 : 0);
        float mx = -3.4e38f;
        for (int u = 0, iu = 0, ju = 0; u < 36; ++u) {
            int ru = (((wh6 + iu) < 45) ? 2 : 0) | (((ww6 + ju) < 45) ? 1 : 0);
            if (ru == rv) mx = fmaxf(mx, sc[v * 48 + u]);
            if (++ju == 6) { ju = 0; ++iu; }
        }
        float sum = 0.0f;
        for (int u = 0, iu = 0, ju = 0; u < 36; ++u) {
            int ru = (((wh6 + iu) < 45) ? 2 : 0) | (((ww6 + ju) < 45) ? 1 : 0);
            float p = (ru == rv) ? __expf(sc[v * 48 + u] - mx) : 0.0f;
            sc[v * 48 + u] = p;
            sum += p;
            if (++ju == 6) { ju = 0; ++iu; }
        }
        float inv = 1.0f / sum;
        for (int u = 0; u < 36; ++u) at[v * 48 + u] = (bf16)(sc[v * 48 + u] * inv);
        for (int u = 36; u < 48; ++u) at[v * 48 + u] = (bf16)0.0f;
    }
    __syncthreads();

    // out = attn @ v  (K = 36 valid; attn cols >=36 are zero, vT guarded)
    const size_t ybase = (size_t)mw * 36 * 384 + h * 48;
    for (int mi = 0; mi < 3; ++mi) {
        v16bf a0 = fragA(at, 48, mi * 16, 0, 48);
        v16bf a1 = fragA(at, 48, mi * 16, 32, 48);
        for (int ni = 0; ni < 3; ++ni) {
            AccF acc; zero_acc(acc);
            acc.v = wmma_bf16(a0, fragB(vt, 48, ni * 16, 0, 36), acc.v);
            acc.v = wmma_bf16(a1, fragB(vt, 48, ni * 16, 32, 36), acc.v);
            for (int i = 0; i < 8; ++i) {
                int v = mi * 16 + mh + i;
                int d = ni * 16 + nl;
                if (v < 36) yg[ybase + (size_t)v * 384 + d] = (bf16)acc.e[i];
            }
        }
    }
}

// ---------------- kernel 3: proj GEMM + window-reverse + un-roll ----------------
__global__ __launch_bounds__(128) void proj_kernel(
    const bf16* __restrict__ yg, const bf16* __restrict__ wp, const float* __restrict__ bp,
    float* __restrict__ out) {
    __shared__ __align__(16) bf16 As[2][64 * 32];
    __shared__ __align__(16) bf16 Bs[2][64 * 32];
    const int tid = threadIdx.x;
    const int lane = tid & 31, wave = tid >> 5;
    const int m0 = blockIdx.x * 64, n0 = blockIdx.y * 64;
    const int wm = (wave >> 1) * 32, wn = (wave & 1) * 32;

    const bf16* asrc[2];
    int aoff[2];
#pragma unroll
    for (int t = 0; t < 2; ++t) {
        int c = tid + t * 128;                 // 0..255
        int row = c >> 2, cp = (c & 3) << 3;
        asrc[t] = yg + (size_t)(m0 + row) * 384 + cp;
        aoff[t] = row * 32 + cp;
    }
    const bf16* bsrc[2];
    int bn[2], bk[2];
#pragma unroll
    for (int t = 0; t < 2; ++t) {
        int c = tid + t * 128;
        int kr = c >> 3, np = (c & 7) << 3;
        bsrc[t] = wp + (size_t)kr * 384 + n0 + np;
        bn[t] = np; bk[t] = kr;
    }

    auto fill = [&](int buf, int k0) {
        uint4 pa[2], pb[2];
#pragma unroll
        for (int t = 0; t < 2; ++t) {
            const bf16* src = asrc[t] + k0;
            if (k0 + 32 < 384) __builtin_prefetch(src + 32, 0, 0);
            pa[t] = *(const uint4*)src;
        }
#pragma unroll
        for (int t = 0; t < 2; ++t) pb[t] = *(const uint4*)(bsrc[t] + (size_t)k0 * 384);
#pragma unroll
        for (int t = 0; t < 2; ++t) *(uint4*)(&As[buf][aoff[t]]) = pa[t];
#pragma unroll
        for (int t = 0; t < 2; ++t) {
            const bf16* pe = (const bf16*)&pb[t];
            for (int e = 0; e < 8; ++e) Bs[buf][(bn[t] + e) * 32 + bk[t]] = pe[e];
        }
    };

    AccF acc[2][2];
    zero_acc(acc[0][0]); zero_acc(acc[0][1]); zero_acc(acc[1][0]); zero_acc(acc[1][1]);

    fill(0, 0);
    __syncthreads();
    for (int kt = 0; kt < 12; ++kt) {
        int cur = kt & 1;
        if (kt < 11) fill(cur ^ 1, (kt + 1) * 32);
        v16bf bfr[2];
        bfr[0] = fragB(Bs[cur], 32, wn, 0, 32);
        bfr[1] = fragB(Bs[cur], 32, wn + 16, 0, 32);
        for (int fm = 0; fm < 2; ++fm) {
            v16bf a = fragA(As[cur], 32, wm + fm * 16, 0, 32);
            acc[fm][0].v = wmma_bf16(a, bfr[0], acc[fm][0].v);
            acc[fm][1].v = wmma_bf16(a, bfr[1], acc[fm][1].v);
        }
        __syncthreads();
    }

    const int nl = lane & 15, mh = (lane >> 4) << 3;
    for (int fm = 0; fm < 2; ++fm)
        for (int fn = 0; fn < 2; ++fn)
            for (int i = 0; i < 8; ++i) {
                int m = m0 + wm + fm * 16 + mh + i;
                int n = n0 + wn + fn * 16 + nl;
                float val = acc[fm][fn].e[i] + bp[n];
                int mw = m / 36, vv = m - mw * 36;
                int b = mw >> 6, wiw = mw & 63;
                int hr = (wiw >> 3) * 6 + vv / 6;
                int wr = (wiw & 7) * 6 + vv % 6;
                int hs = hr + 3; if (hs >= 48) hs -= 48;
                int wd = wr + 3; if (wd >= 48) wd -= 48;
                out[((size_t)(b * 48 + hs) * 48 + wd) * 384 + n] = val;
            }
}

extern "C" void kernel_launch(void* const* d_in, const int* in_sizes, int n_in,
                              void* d_out, int out_size, void* d_ws, size_t ws_size,
                              hipStream_t stream) {
    const float* x      = (const float*)d_in[0];
    const float* w_qkv  = (const float*)d_in[1];
    const float* b_qkv  = (const float*)d_in[2];
    const float* w_proj = (const float*)d_in[3];
    const float* b_proj = (const float*)d_in[4];
    float* out = (float*)d_out;

    char* ws = (char*)d_ws;
    bf16* wqb = (bf16*)(ws + 0);          //   884,736 B : wqkv bf16
    bf16* wpb = (bf16*)(ws + 884736);     //   294,912 B : wproj bf16
    bf16* qg  = (bf16*)(ws + 1179648);    // 28,311,552 B : q  (2048,8,36,48)
    bf16* kg  = (bf16*)(ws + 29491200);   // 28,311,552 B : k  (2048,8,36,48)
    bf16* vtg = (bf16*)(ws + 57802752);   // 28,311,552 B : vT (2048,8,48,36)
    bf16* yg  = (bf16*)(ws + 86114304);   // 56,623,104 B : attn out (73728,384)

    cvt_kernel<<<1728, 256, 0, stream>>>(w_qkv, w_proj, wqb, wpb);
    qkv_kernel<<<dim3(1152, 18), 128, 0, stream>>>(x, wqb, b_qkv, qg, kg, vtg);
    attn_kernel<<<dim3(2048, 4), 64, 0, stream>>>(qg, kg, vtg, yg);
    proj_kernel<<<dim3(1152, 6), 128, 0, stream>>>(yg, wpb, b_proj, out);
}
